// ConvLSTMEncoder_75574244540515
// MI455X (gfx1250) — compile-verified, matrix operand order checked by reading, not run on hardware
//
#include <hip/hip_runtime.h>

// ---------------------------------------------------------------------------
// ConvLSTMEncoder for MI455X (gfx1250): bf16 WMMA everywhere, persistent
// LSTM scan with Wh resident in LDS (fragment order), TDM double-buffered
// prefetch of the gate pre-activations, grid barrier per step.
// ---------------------------------------------------------------------------

typedef __bf16 bf16;
typedef bf16  v16bf __attribute__((ext_vector_type(16)));
typedef bf16  v8bf  __attribute__((ext_vector_type(8)));
typedef float v8f   __attribute__((ext_vector_type(8)));
typedef unsigned int u32;
typedef u32 u32x4 __attribute__((ext_vector_type(4)));
typedef int  i32x4 __attribute__((ext_vector_type(4)));
typedef int  i32x8 __attribute__((ext_vector_type(8)));

#define HID 512
#define XD  128
#define TLEN 1024
#define BATCH 64
#define MTOT (BATCH * TLEN)   // 65536 rows (b*T + t)
#define GHID (4 * HID)        // 2048 gate columns
#define LSTM_WGS 32           // persistent workgroups for the scan
#define TPAD (TLEN + 2)       // padded time extent for conv input

#if __has_builtin(__builtin_amdgcn_tensor_load_to_lds)
#define USE_TDM 1
#else
#define USE_TDM 0
#endif

// --------------------------- WMMA helpers ----------------------------------

__device__ __forceinline__ v8f wmma_bf16(v16bf a, v16bf b, v8f c) {
  // D = A(16x32) * B(32x16) + C, f32 accumulate
  return __builtin_amdgcn_wmma_f32_16x16x32_bf16(
      /*neg_a=*/false, a, /*neg_b=*/false, b,
      /*c_mod=*/(short)0, c, /*reuse_a=*/false, /*reuse_b=*/false);
}

// A fragment (16x32, MxK) from row-major bf16 matrix.
// lane<16: row=lane, K = {k0..k0+7, k0+16..k0+23}
// lane>=16: row=lane-16, K = {k0+8..k0+15, k0+24..k0+31}
__device__ __forceinline__ v16bf load_a_bf16(const bf16* __restrict__ A,
                                             int lda, int row0, int k0,
                                             int lane) {
  const bf16* p = A + (size_t)(row0 + (lane & 15)) * lda + k0 + ((lane >> 4) << 3);
  v8bf lo = *(const v8bf*)p;
  v8bf hi = *(const v8bf*)(p + 16);
  return __builtin_shufflevector(lo, hi, 0, 1, 2, 3, 4, 5, 6, 7,
                                 8, 9, 10, 11, 12, 13, 14, 15);
}

// B fragment (32x16, KxN) from LDS already laid out in fragment order:
// 512 halfs per fragment, lane-major, 32B contiguous per lane -> ds_load_b128 x2
__device__ __forceinline__ v16bf load_b_lds(const bf16* __restrict__ lds,
                                            int frag, int lane) {
  return *(const v16bf*)(lds + (size_t)frag * 512 + lane * 16);
}

// Stage f32 row-major B[k][n] (ldb cols) into LDS fragment order (one-time).
// frag = kt*nTiles + nt ; n = n0 + nt*nStride + (lane&15)
__device__ __forceinline__ void stage_b_f32(const float* __restrict__ B,
                                            int ldb, int n0, int kTiles,
                                            int nTiles, int nStride,
                                            bf16* __restrict__ lds) {
  int total = kTiles * nTiles * 512;
  for (int idx = threadIdx.x; idx < total; idx += blockDim.x) {
    int frag = idx >> 9;
    int rem  = idx & 511;
    int l = rem >> 4;
    int h = rem & 15;
    int kt = frag / nTiles;
    int nt = frag - kt * nTiles;
    int k = kt * 32 + ((l >> 4) << 3) + (h < 8 ? h : h + 8);
    int n = n0 + nt * nStride + (l & 15);
    lds[idx] = (bf16)B[(size_t)k * ldb + n];
  }
}

// --------------------------- grid barrier ----------------------------------

__device__ __forceinline__ void grid_barrier(unsigned* cnt,
                                             volatile unsigned* gen,
                                             unsigned nwg) {
  __syncthreads();
  if (threadIdx.x == 0) {
    __threadfence();
    unsigned g = *gen;
    if (atomicAdd(cnt, 1u) == nwg - 1) {
      *cnt = 0;
      __threadfence();
      *gen = g + 1;
    } else {
      while (*gen == g) { __builtin_amdgcn_s_sleep(2); }
    }
    __threadfence();
  }
  __syncthreads();
}

__global__ void barrier_init_kernel(unsigned* p) { p[0] = 0u; p[1] = 0u; }

// ----------------- Kernel 0: pad + convert x -> bf16 -----------------------
// xpad[b][tr][xd], tr in [0, T+2), tr = t+1; boundary rows zero (SAME pad).

__global__ void pad_convert_kernel(const float* __restrict__ x,
                                   bf16* __restrict__ xpad) {
  int idx = blockIdx.x * blockDim.x + threadIdx.x;
  int total = BATCH * TPAD * XD;
  if (idx >= total) return;
  int b  = idx / (TPAD * XD);
  int rm = idx - b * (TPAD * XD);
  int tr = rm / XD;
  int xd = rm - tr * XD;
  int t  = tr - 1;
  float v = (t >= 0 && t < TLEN) ? x[((size_t)(b * TLEN + t)) * XD + xd] : 0.f;
  xpad[idx] = (bf16)v;
}

// --------------------------- Kernel 1: Conv1D ------------------------------
// conv[m][n] = sum_{kc<384} Aconv[m][kc] * conv_w[kc][n] + conv_b[n]
// Aconv row m=(b,t) is concat_k xpad[b][t+k][:]; branch-free bf16 loads.

__device__ __forceinline__ v16bf load_a_conv(const bf16* __restrict__ xp,
                                             int row, int kb, int lane) {
  int r = row + (lane & 15);
  int b = r >> 10;
  int t = r & 1023;
  int k0 = kb + ((lane >> 4) << 3);
  const bf16* base = xp + (size_t)b * (TPAD * XD);
  int c0 = k0;           // first 8-elem chunk within the 384 concat
  int c1 = k0 + 16;      // second chunk
  v8bf lo = *(const v8bf*)(base + (t + (c0 >> 7)) * XD + (c0 & 127));
  v8bf hi = *(const v8bf*)(base + (t + (c1 >> 7)) * XD + (c1 & 127));
  return __builtin_shufflevector(lo, hi, 0, 1, 2, 3, 4, 5, 6, 7,
                                 8, 9, 10, 11, 12, 13, 14, 15);
}

__global__ void conv_kernel(const bf16* __restrict__ xp,
                            const float* __restrict__ cw,
                            const float* __restrict__ cb,
                            bf16* __restrict__ out) {
  extern __shared__ char smem[];
  bf16* bl = (bf16*)smem;
  stage_b_f32(cw, HID, blockIdx.y * 64, /*kTiles=*/12, /*nTiles=*/4, 16, bl);
  __syncthreads();

  int lane = threadIdx.x & 31;
  int w    = threadIdx.x >> 5;
  int nt = w & 3, mh = w >> 2;
  int row0 = blockIdx.x * 64 + mh * 32;

  v8f acc0 = {}, acc1 = {};
#pragma unroll 4
  for (int kt = 0; kt < 12; ++kt) {
    int kb = kt * 32;
    v16bf a0 = load_a_conv(xp, row0, kb, lane);
    v16bf a1 = load_a_conv(xp, row0 + 16, kb, lane);
    v16bf bf = load_b_lds(bl, kt * 4 + nt, lane);
    acc0 = wmma_bf16(a0, bf, acc0);
    acc1 = wmma_bf16(a1, bf, acc1);
  }

  int n = blockIdx.y * 64 + nt * 16 + (lane & 15);
  float bias = cb[n];
  int mb = row0 + ((lane >> 4) << 3);
#pragma unroll
  for (int r = 0; r < 8; ++r) {
    out[(size_t)(mb + r) * HID + n]      = (bf16)(acc0[r] + bias);
    out[(size_t)(mb + 16 + r) * HID + n] = (bf16)(acc1[r] + bias);
  }
}

// --------------------- Kernel 2: xz = conv @ Wx + b ------------------------
// (65536x512)·(512x2048), f32 output (feeds the scan's gate pre-activation).

__global__ void xw_kernel(const bf16* __restrict__ A,
                          const float* __restrict__ Wx,
                          const float* __restrict__ bg,
                          float* __restrict__ xz) {
  extern __shared__ char smem[];
  bf16* bl = (bf16*)smem;
  stage_b_f32(Wx, GHID, blockIdx.y * 64, /*kTiles=*/16, /*nTiles=*/4, 16, bl);
  __syncthreads();

  int lane = threadIdx.x & 31;
  int w    = threadIdx.x >> 5;
  int nt = w & 3, mh = w >> 2;
  int row0 = blockIdx.x * 64 + mh * 32;

  v8f acc0 = {}, acc1 = {};
#pragma unroll 4
  for (int kt = 0; kt < 16; ++kt) {
    v16bf a0 = load_a_bf16(A, HID, row0, kt * 32, lane);
    v16bf a1 = load_a_bf16(A, HID, row0 + 16, kt * 32, lane);
    v16bf bf = load_b_lds(bl, kt * 4 + nt, lane);
    acc0 = wmma_bf16(a0, bf, acc0);
    acc1 = wmma_bf16(a1, bf, acc1);
  }

  int n = blockIdx.y * 64 + nt * 16 + (lane & 15);
  float bias = bg[n];
  int mb = row0 + ((lane >> 4) << 3);
#pragma unroll
  for (int r = 0; r < 8; ++r) {
    xz[(size_t)(mb + r) * GHID + n]      = acc0[r] + bias;
    xz[(size_t)(mb + 16 + r) * GHID + n] = acc1[r] + bias;
  }
}

// --------------------- TDM: 3D gate-slice prefetch -------------------------
// Tile: x = 16 floats (hid slice), y = 4 gates (stride 512), z = 64 batches
// (stride T*4H). One descriptor moves the whole 16KB per-step slice to LDS.

#if USE_TDM
__device__ __forceinline__ void tdm_load_xz_tile(const float* __restrict__ gsrc,
                                                 void* ldst) {
  unsigned long long ga = (unsigned long long)(size_t)gsrc;
  unsigned lds = (unsigned)(size_t)ldst;
  u32x4 g0;
  g0[0] = 1u;                                         // count=1 valid descriptor
  g0[1] = lds;                                        // lds_addr (bytes)
  g0[2] = (u32)(ga & 0xffffffffu);                    // global_addr lo
  g0[3] = (u32)((ga >> 32) & 0x01ffffffu) | (2u << 30); // addr hi | type=2
  const u32 dim0 = 1u << 20, dim1 = 1u << 20;         // generous OOB bounds
  const u32 d0s = 512u;                               // y stride (gate)
  const u32 d1s = (u32)TLEN * (u32)GHID;              // z stride (batch)
  i32x8 g1;
  g1[0] = (int)(2u << 16);                            // data_size = 4B
  g1[1] = (int)((dim0 & 0xffffu) << 16);              // tensor_dim0 lo
  g1[2] = (int)((dim0 >> 16) | ((dim1 & 0xffffu) << 16)); // dim0 hi | dim1 lo
  g1[3] = (int)((dim1 >> 16) | (16u << 16));          // dim1 hi | tile_dim0=16
  g1[4] = (int)(4u | (64u << 16));                    // tile_dim1=4 | tile_dim2=64
  g1[5] = (int)d0s;                                   // dim0_stride lo
  g1[6] = (int)((d1s & 0xffffu) << 16);               // dim0_stride hi | dim1_stride lo
  g1[7] = (int)(d1s >> 16);                           // dim1_stride hi
  i32x4 g2;
  g2[0] = (int)(1u << 20);                            // tensor_dim2
  g2[1] = 0; g2[2] = 0; g2[3] = 0;
  i32x4 g3; g3[0] = 0; g3[1] = 0; g3[2] = 0; g3[3] = 0;
#if defined(__clang_major__) && __clang_major__ >= 23
  i32x8 g4 = {};
  __builtin_amdgcn_tensor_load_to_lds(g0, g1, g2, g3, g4, 0);
#else
  __builtin_amdgcn_tensor_load_to_lds(g0, g1, g2, g3, 0);
#endif
}
#endif

// ------------------- Kernel 3: persistent LSTM scan ------------------------
// 32 persistent WGs; WG nb owns hid slice [nb*16, nb*16+16) for all 4 gates.
// LDS: Wh fragments 64KB | z tile 16KB | xz double buffer 2x16KB = 112KB.
// One grid barrier per timestep; c-state in registers; h double-buffered bf16.

__global__ void lstm_kernel(const float* __restrict__ Wh,
                            const float* __restrict__ xz,
                            bf16* __restrict__ hseq,
                            bf16* __restrict__ hbuf,
                            unsigned* __restrict__ cnt,
                            volatile unsigned* __restrict__ gen) {
  extern __shared__ char smem[];
  bf16*  bl   = (bf16*)smem;                     // 64 frags * 512 halfs = 64KB
  float* zl   = (float*)(smem + 65536);          // 64x64 f32 = 16KB
  float* xzl0 = (float*)(smem + 65536 + 16384);  // [64][4][16] f32 = 16KB
  float* xzl1 = xzl0 + 4096;                     // second buffer

  int nb = blockIdx.x;
  // Stage Wh fragments: frag = kt*4 + gate ; n = gate*512 + nb*16 + (l&15)
  stage_b_f32(Wh, GHID, nb * 16, /*kTiles=*/16, /*nTiles=*/4, /*nStride=*/HID, bl);

  // Zero h0 (buffer 0): this block's 1/32 slice of 64x512 bf16
  for (int i = threadIdx.x; i < (BATCH * HID) / LSTM_WGS; i += blockDim.x)
    hbuf[nb * ((BATCH * HID) / LSTM_WGS) + i] = (bf16)0.f;

  int lane = threadIdx.x & 31;
  int w    = threadIdx.x >> 5;
  int gate = w & 3, mh = w >> 2;

  float creg[4] = {0.f, 0.f, 0.f, 0.f};

#if USE_TDM
  if (w == 0) tdm_load_xz_tile(xz + (size_t)nb * 16, xzl0);  // step 0 slice
#endif

  grid_barrier(cnt, gen, LSTM_WGS);

  for (int t = 0; t < TLEN; ++t) {
#if USE_TDM
    if (t + 1 < TLEN && w == 0)
      tdm_load_xz_tile(xz + (size_t)(t + 1) * GHID + nb * 16,
                       ((t + 1) & 1) ? xzl1 : xzl0);
#endif

    const bf16* hr = hbuf + (size_t)(t & 1) * (BATCH * HID);
    bf16*       hw = hbuf + (size_t)((t & 1) ^ 1) * (BATCH * HID);

    // z_tile(64x64) = h(64x512) @ Wh_slice(512x64), f32 accumulate
    v8f acc0 = {}, acc1 = {};
#pragma unroll 4
    for (int kt = 0; kt < 16; ++kt) {
      v16bf a0 = load_a_bf16(hr, HID, mh * 32, kt * 32, lane);
      v16bf a1 = load_a_bf16(hr, HID, mh * 32 + 16, kt * 32, lane);
      v16bf bf = load_b_lds(bl, kt * 4 + gate, lane);
      acc0 = wmma_bf16(a0, bf, acc0);
      acc1 = wmma_bf16(a1, bf, acc1);
    }

    int col = gate * 16 + (lane & 15);
    int mb  = mh * 32 + ((lane >> 4) << 3);
#pragma unroll
    for (int r = 0; r < 8; ++r) {
      zl[(mb + r) * 64 + col]      = acc0[r];
      zl[(mb + 16 + r) * 64 + col] = acc1[r];
    }

#if USE_TDM
    if (w == 0) {
      if (t + 1 < TLEN) __builtin_amdgcn_s_wait_tensorcnt(1);  // step-t op done
      else              __builtin_amdgcn_s_wait_tensorcnt(0);
    }
#endif
    __syncthreads();

#if USE_TDM
    const float* xzc = (t & 1) ? xzl1 : xzl0;
#endif
    // Gate math: thread owns 4 fixed (batch, hid) elements; c in registers.
#pragma unroll
    for (int j = 0; j < 4; ++j) {
      int e  = threadIdx.x + 256 * j;   // 0..1023
      int bI = e >> 4;                  // batch row 0..63
      int hh = e & 15;                  // hid within slice
#if USE_TDM
      float zi = zl[bI * 64 + hh]      + xzc[(bI * 4 + 0) * 16 + hh];
      float zf = zl[bI * 64 + 16 + hh] + xzc[(bI * 4 + 1) * 16 + hh];
      float zg = zl[bI * 64 + 32 + hh] + xzc[(bI * 4 + 2) * 16 + hh];
      float zo = zl[bI * 64 + 48 + hh] + xzc[(bI * 4 + 3) * 16 + hh];
#else
      size_t xo = ((size_t)(bI * TLEN + t)) * GHID + nb * 16 + hh;
      float zi = zl[bI * 64 + hh]      + xz[xo];
      float zf = zl[bI * 64 + 16 + hh] + xz[xo + 512];
      float zg = zl[bI * 64 + 32 + hh] + xz[xo + 1024];
      float zo = zl[bI * 64 + 48 + hh] + xz[xo + 1536];
#endif
      float ig = 1.f / (1.f + __expf(-zi));
      float fg = 1.f / (1.f + __expf(-zf));
      float gg = tanhf(zg);
      float og = 1.f / (1.f + __expf(-zo));
      creg[j] = fg * creg[j] + ig * gg;
      float hn = og * tanhf(creg[j]);
      int hid = nb * 16 + hh;
      hw[bI * HID + hid] = (bf16)hn;
      hseq[((size_t)(bI * TLEN + t)) * HID + hid] = (bf16)hn;
    }
    grid_barrier(cnt, gen, LSTM_WGS);
  }
}

// ------------------- Kernel 4a: act1 = relu(hseq@W1 + b1) ------------------

__global__ void head1_kernel(const bf16* __restrict__ A,
                             const float* __restrict__ W1,
                             const float* __restrict__ b1,
                             bf16* __restrict__ act1) {
  extern __shared__ char smem[];
  bf16* bl = (bf16*)smem;
  stage_b_f32(W1, HID, blockIdx.y * 64, /*kTiles=*/16, /*nTiles=*/4, 16, bl);
  __syncthreads();

  int lane = threadIdx.x & 31;
  int w    = threadIdx.x >> 5;
  int nt = w & 3, mh = w >> 2;
  int row0 = blockIdx.x * 64 + mh * 32;

  v8f acc0 = {}, acc1 = {};
#pragma unroll 4
  for (int kt = 0; kt < 16; ++kt) {
    v16bf a0 = load_a_bf16(A, HID, row0, kt * 32, lane);
    v16bf a1 = load_a_bf16(A, HID, row0 + 16, kt * 32, lane);
    v16bf bf = load_b_lds(bl, kt * 4 + nt, lane);
    acc0 = wmma_bf16(a0, bf, acc0);
    acc1 = wmma_bf16(a1, bf, acc1);
  }

  int n = blockIdx.y * 64 + nt * 16 + (lane & 15);
  float bias = b1[n];
  int mb = row0 + ((lane >> 4) << 3);
#pragma unroll
  for (int r = 0; r < 8; ++r) {
    act1[(size_t)(mb + r) * HID + n]      = (bf16)fmaxf(acc0[r] + bias, 0.f);
    act1[(size_t)(mb + 16 + r) * HID + n] = (bf16)fmaxf(acc1[r] + bias, 0.f);
  }
}

// -------- Kernel 4b: out = act1@W2 + b2 ; split -> mu | log_sigma ----------

__global__ void head2_kernel(const bf16* __restrict__ A,
                             const float* __restrict__ W2,
                             const float* __restrict__ b2,
                             float* __restrict__ outp) {
  extern __shared__ char smem[];
  bf16* bl = (bf16*)smem;
  stage_b_f32(W2, 128, blockIdx.y * 64, /*kTiles=*/16, /*nTiles=*/4, 16, bl);
  __syncthreads();

  int lane = threadIdx.x & 31;
  int w    = threadIdx.x >> 5;
  int nt = w & 3, mh = w >> 2;
  int row0 = blockIdx.x * 64 + mh * 32;

  v8f acc0 = {}, acc1 = {};
#pragma unroll 4
  for (int kt = 0; kt < 16; ++kt) {
    v16bf a0 = load_a_bf16(A, HID, row0, kt * 32, lane);
    v16bf a1 = load_a_bf16(A, HID, row0 + 16, kt * 32, lane);
    v16bf bf = load_b_lds(bl, kt * 4 + nt, lane);
    acc0 = wmma_bf16(a0, bf, acc0);
    acc1 = wmma_bf16(a1, bf, acc1);
  }

  int n = blockIdx.y * 64 + nt * 16 + (lane & 15);  // 0..127
  float bias = b2[n];
  size_t half_off = (size_t)MTOT * 64;               // log_sigma offset
  int mb = row0 + ((lane >> 4) << 3);
#pragma unroll
  for (int r = 0; r < 8; ++r) {
    float v0 = acc0[r] + bias;
    float v1 = acc1[r] + bias;
    int m0 = mb + r, m1 = mb + 16 + r;
    if (n < 64) {
      outp[(size_t)m0 * 64 + n] = v0;
      outp[(size_t)m1 * 64 + n] = v1;
    } else {
      outp[half_off + (size_t)m0 * 64 + (n - 64)] = v0;
      outp[half_off + (size_t)m1 * 64 + (n - 64)] = v1;
    }
  }
}

// ------------------------------- launcher ----------------------------------

extern "C" void kernel_launch(void* const* d_in, const int* in_sizes, int n_in,
                              void* d_out, int out_size, void* d_ws,
                              size_t ws_size, hipStream_t stream) {
  const float* x   = (const float*)d_in[0];  // (B,T,XD)
  const float* cw  = (const float*)d_in[1];  // (3,128,512) == (384,512)
  const float* cb  = (const float*)d_in[2];  // (512)
  const float* Wx  = (const float*)d_in[3];  // (512,2048)
  const float* Wh  = (const float*)d_in[4];  // (512,2048)
  const float* bg  = (const float*)d_in[5];  // (2048)
  const float* W1  = (const float*)d_in[6];  // (512,512)
  const float* b1  = (const float*)d_in[7];  // (512)
  const float* W2  = (const float*)d_in[8];  // (512,128)
  const float* b2  = (const float*)d_in[9];  // (128)

  char* ws = (char*)d_ws;
  size_t off = 0;
  bf16* xpad   = (bf16*)(ws + off); off += (size_t)BATCH * TPAD * XD * 2; // ~16.8MB
  off = (off + 255) & ~(size_t)255;
  bf16* convbf = (bf16*)(ws + off); off += (size_t)MTOT * HID * 2;        // 64MB
  float* xz    = (float*)(ws + off); off += (size_t)MTOT * GHID * 4;      // 512MB
  bf16* hseq   = (bf16*)(ws + off); off += (size_t)MTOT * HID * 2;        // 64MB
  bf16* hbuf   = (bf16*)(ws + off); off += (size_t)2 * BATCH * HID * 2;   // 128KB
  unsigned* bar = (unsigned*)(ws + off); off += 256;
  bf16* act1 = convbf;  // conv activations dead after xw_kernel -> reuse

  barrier_init_kernel<<<1, 1, 0, stream>>>(bar);

  // Pad + convert x to bf16 once (keeps conv GEMM inner loop branch-free)
  {
    int total = BATCH * TPAD * XD;
    pad_convert_kernel<<<(total + 255) / 256, 256, 0, stream>>>(x, xpad);
  }

  // Conv1D as GEMM: M=65536, K=384, N=512
  conv_kernel<<<dim3(MTOT / 64, HID / 64), 256, 12 * 4 * 512 * 2, stream>>>(
      xpad, cw, cb, convbf);

  // Input projection: M=65536, K=512, N=2048 (hoisted out of the scan)
  xw_kernel<<<dim3(MTOT / 64, GHID / 64), 256, 16 * 4 * 512 * 2, stream>>>(
      convbf, Wx, bg, xz);

  // Persistent LSTM scan: 32 WGs, 112KB dynamic LDS each
  lstm_kernel<<<dim3(LSTM_WGS), 256,
                16 * 4 * 512 * 2 + 64 * 64 * 4 + 2 * 4096 * 4, stream>>>(
      Wh, xz, hseq, hbuf, bar, bar + 1);

  // Head: relu(hseq@W1+b1) then @W2+b2 with mu/log_sigma split
  head1_kernel<<<dim3(MTOT / 64, HID / 64), 256, 16 * 4 * 512 * 2, stream>>>(
      hseq, W1, b1, act1);
  head2_kernel<<<dim3(MTOT / 64, 128 / 64), 256, 16 * 4 * 512 * 2, stream>>>(
      act1, W2, b2, (float*)d_out);
}
